// GroupQuerySelfAttention_6811818131465
// MI455X (gfx1250) — matrix operand order repaired and checked by reading, not compiled, not run on hardware
//
#include <hip/hip_runtime.h>
#include <hip/hip_bf16.h>

// ---------------------------------------------------------------------------
// Types for CDNA5 WMMA (wave32): v_wmma_f32_16x16x32_bf16
// ---------------------------------------------------------------------------
typedef __bf16 bf16_t;
typedef bf16_t bf16x8 __attribute__((ext_vector_type(8)));
typedef bf16_t v16bf  __attribute__((ext_vector_type(16)));
typedef float  v8f    __attribute__((ext_vector_type(8)));

#define B_  2
#define S_  1024
#define D_  2048
#define DKV 512
#define H_  32
#define HKV 8
#define DH  64

__device__ __forceinline__ v8f wmma_bf16(v16bf a, v16bf b, v8f c) {
  // D = A(16x32 bf16) x B(32x16 bf16) + C(16x16 f32)
  return __builtin_amdgcn_wmma_f32_16x16x32_bf16(
      /*neg_a=*/false, a, /*neg_b=*/false, b,
      /*c_mod=*/(short)0, c, /*reuse_a=*/false, /*reuse_b=*/false);
}

// A fragment (16x32, row-major source, lda elements):
//   lane m = lane&15 ; slots 0..7  -> K = k0 + 8*(lane>>4) + slot
//                      slots 8..15 -> K = k0 + 16 + 8*(lane>>4) + (slot-8)
__device__ __forceinline__ v16bf load_frag_a(const bf16_t* base, int row0,
                                             int lda, int k0, int lane) {
  int m  = lane & 15;
  int kk = k0 + 8 * (lane >> 4);
  const bf16_t* p = base + (size_t)(row0 + m) * lda + kk;
  bf16x8 lo = *(const bf16x8*)(p);
  bf16x8 hi = *(const bf16x8*)(p + 16);
  return __builtin_shufflevector(lo, hi, 0,1,2,3,4,5,6,7,8,9,10,11,12,13,14,15);
}

// B fragment (32x16, B[k,n] = W[n0+n, k0+k], W row-major with ldb elements):
//   lane n = lane&15 ; slots 0..7  -> K = k0 + 8*(lane>>4) + slot
//                      slots 8..15 -> K = k0 + 16 + 8*(lane>>4) + (slot-8)
__device__ __forceinline__ v16bf load_frag_b(const bf16_t* base, int n0,
                                             int ldb, int k0, int lane) {
  int n  = lane & 15;
  int kk = k0 + 8 * (lane >> 4);
  const bf16_t* p = base + (size_t)(n0 + n) * ldb + kk;
  bf16x8 lo = *(const bf16x8*)(p);
  bf16x8 hi = *(const bf16x8*)(p + 16);
  return __builtin_shufflevector(lo, hi, 0,1,2,3,4,5,6,7,8,9,10,11,12,13,14,15);
}

// ---------------------------------------------------------------------------
// f32 -> bf16 flat convert
// ---------------------------------------------------------------------------
__global__ __launch_bounds__(256) void cvt_bf16_kernel(const float* __restrict__ in,
                                                       bf16_t* __restrict__ out, int n) {
  int i = blockIdx.x * blockDim.x + threadIdx.x;
  if (i < n) out[i] = (bf16_t)in[i];
}

// ---------------------------------------------------------------------------
// Generic GEMM: C[M,N] (f32) = A[M,K] (bf16, row-major) x W[N,K]^T (bf16)
// One wave computes a 32x64 tile (2 M-frags x 4 N-frags): per k-step
// 12 b128 loads feed 8 WMMAs (B fragments amortized over two M tiles).
// ---------------------------------------------------------------------------
__global__ __launch_bounds__(256) void gemm_bf16_kernel(const bf16_t* __restrict__ A,
                                                        const bf16_t* __restrict__ W,
                                                        float* __restrict__ C,
                                                        int M, int N, int K) {
  int lane = threadIdx.x & 31;
  int wave = (blockIdx.x * blockDim.x + threadIdx.x) >> 5;
  int tilesN = N >> 6;
  int mT = wave / tilesN;
  int nT = wave - mT * tilesN;
  int m0 = mT << 5;                      // 32 rows per wave
  int n0 = nT << 6;                      // 64 cols per wave
  if (m0 >= M) return;

  v8f acc0[4] = {}, acc1[4] = {};
  for (int k0 = 0; k0 < K; k0 += 32) {
    v16bf a0 = load_frag_a(A, m0,      K, k0, lane);
    v16bf a1 = load_frag_a(A, m0 + 16, K, k0, lane);
#pragma unroll
    for (int t = 0; t < 4; ++t) {
      v16bf b = load_frag_b(W, n0 + t * 16, K, k0, lane);
      acc0[t] = wmma_bf16(a0, b, acc0[t]);
      acc1[t] = wmma_bf16(a1, b, acc1[t]);
    }
  }
  int half = lane >> 4, col = lane & 15;
#pragma unroll
  for (int t = 0; t < 4; ++t)
#pragma unroll
    for (int r = 0; r < 8; ++r) {
      C[(size_t)(m0 + r + 8 * half) * N + n0 + t * 16 + col]      = acc0[t][r];
      C[(size_t)(m0 + 16 + r + 8 * half) * N + n0 + t * 16 + col] = acc1[t][r];
    }
}

// ---------------------------------------------------------------------------
// RoPE + repack Q: Qf[b,s, h*64+dh] f32 -> Qb[b,h,s,dh] bf16
// ---------------------------------------------------------------------------
__global__ __launch_bounds__(256) void rope_pack_q_kernel(const float* __restrict__ Qf,
                                                          const int* __restrict__ tok,
                                                          bf16_t* __restrict__ Qb) {
  int idx = blockIdx.x * blockDim.x + threadIdx.x;   // pair index, D_/2 per row
  if (idx >= B_ * S_ * (D_ / 2)) return;
  int dpair = idx & (D_ / 2 - 1);
  int row   = idx >> 10;                              // b*S + s
  int s = row & (S_ - 1);
  int b = row >> 10;
  int d0 = dpair * 2;
  int h  = d0 >> 6;
  int dd = d0 & 63;                                   // even index within head
  float x1 = Qf[(size_t)row * D_ + d0];
  float x2 = Qf[(size_t)row * D_ + d0 + 1];
  float inv = __expf(-(float)dd * (9.21034037198f / 64.0f));  // theta^{-dd/64}
  float ang = (float)tok[s] * inv;
  float sn, cs;
  __sincosf(ang, &sn, &cs);
  size_t o = (((size_t)(b * H_ + h)) * S_ + s) * DH + dd;
  Qb[o]     = (bf16_t)(x1 * cs - x2 * sn);
  Qb[o + 1] = (bf16_t)(x1 * sn + x2 * cs);
}

// RoPE + repack K: Kf[b,s, hkv*64+dh] f32 -> Kb[b,hkv,s,dh] bf16
__global__ __launch_bounds__(256) void rope_pack_k_kernel(const float* __restrict__ Kf,
                                                          const int* __restrict__ tok,
                                                          bf16_t* __restrict__ Kb) {
  int idx = blockIdx.x * blockDim.x + threadIdx.x;   // pair index, DKV/2 per row
  if (idx >= B_ * S_ * (DKV / 2)) return;
  int dpair = idx & (DKV / 2 - 1);
  int row   = idx >> 8;
  int s = row & (S_ - 1);
  int b = row >> 10;
  int d0 = dpair * 2;
  int hk = d0 >> 6;
  int dd = d0 & 63;
  float x1 = Kf[(size_t)row * DKV + d0];
  float x2 = Kf[(size_t)row * DKV + d0 + 1];
  float inv = __expf(-(float)dd * (9.21034037198f / 64.0f));
  float ang = (float)tok[s] * inv;
  float sn, cs;
  __sincosf(ang, &sn, &cs);
  size_t o = (((size_t)(b * HKV + hk)) * S_ + s) * DH + dd;
  Kb[o]     = (bf16_t)(x1 * cs - x2 * sn);
  Kb[o + 1] = (bf16_t)(x1 * sn + x2 * cs);
}

// Repack V transposed: Vf[b,s, hkv*64+dh] f32 -> Vt[b,hkv,dh,s] bf16
__global__ __launch_bounds__(256) void pack_v_kernel(const float* __restrict__ Vf,
                                                     bf16_t* __restrict__ Vt) {
  int idx = blockIdx.x * blockDim.x + threadIdx.x;
  if (idx >= B_ * S_ * DKV) return;
  int f   = idx & (DKV - 1);
  int row = idx >> 9;
  int s = row & (S_ - 1);
  int b = row >> 10;
  int hk = f >> 6;
  int dh = f & 63;
  Vt[(((size_t)(b * HKV + hk)) * DH + dh) * S_ + s] = (bf16_t)Vf[idx];
}

// ---------------------------------------------------------------------------
// Flash attention: one wave handles one 16-query strip of one head.
// Scores + PV via v_wmma_f32_16x16x32_bf16; P transposed through LDS.
// Output: Ab[b, s, h*64+dh] bf16 (ready to be A of the final GEMM).
// ---------------------------------------------------------------------------
__global__ __launch_bounds__(256) void attn_kernel(const bf16_t* __restrict__ Qb,
                                                   const bf16_t* __restrict__ Kb,
                                                   const bf16_t* __restrict__ Vt,
                                                   bf16_t* __restrict__ Ab) {
  __shared__ __align__(16) bf16_t pstage[8][16 * 32];
  int lane  = threadIdx.x & 31;
  int wslot = threadIdx.x >> 5;
  int wid = (blockIdx.x * blockDim.x + threadIdx.x) >> 5;
  int qb = wid & 63;            // 64 strips of 16 queries
  int h  = (wid >> 6) & 31;
  int b  = wid >> 11;
  int q0 = qb << 4;
  int hkv = h >> 2;             // G = 4

  const bf16_t* Qh = Qb + ((size_t)(b * H_ + h)) * S_ * DH;
  const bf16_t* Kh = Kb + ((size_t)(b * HKV + hkv)) * S_ * DH;
  const bf16_t* Vh = Vt + ((size_t)(b * HKV + hkv)) * DH * S_;

  // Q strip fragments (Kdim = 64 -> two 16x32 A fragments), loaded once.
  v16bf qa0 = load_frag_a(Qh, q0, DH, 0, lane);
  v16bf qa1 = load_frag_a(Qh, q0, DH, 32, lane);

  v8f o0 = {}, o1 = {}, o2 = {}, o3 = {};
  float m_i[8], l_i[8];
#pragma unroll
  for (int r = 0; r < 8; ++r) { m_i[r] = -__builtin_inff(); l_i[r] = 0.f; }

  int half = lane >> 4, col = lane & 15;
  const float scale = 0.125f;                 // 1/sqrt(64)
  int nblocks = (q0 + 16 + 31) >> 5;          // causal: only key blocks <= strip end
  bf16_t* myP = &pstage[wslot][0];

  for (int kb = 0; kb < nblocks; ++kb) {
    int kbase = kb << 5;
    // ---- scores S = Q K^T (16x32 tile = two 16x16 C frags) ----
    v8f s0 = {}, s1 = {};
    s0 = wmma_bf16(qa0, load_frag_b(Kh, kbase,      DH, 0,  lane), s0);
    s0 = wmma_bf16(qa1, load_frag_b(Kh, kbase,      DH, 32, lane), s0);
    s1 = wmma_bf16(qa0, load_frag_b(Kh, kbase + 16, DH, 0,  lane), s1);
    s1 = wmma_bf16(qa1, load_frag_b(Kh, kbase + 16, DH, 32, lane), s1);

    // ---- scale + causal mask + per-row block max ----
    float mb[8];
#pragma unroll
    for (int r = 0; r < 8; ++r) {
      int qrow = q0 + r + 8 * half;
      float v0 = s0[r] * scale;
      float v1 = s1[r] * scale;
      if (kbase + col > qrow)      v0 = -__builtin_inff();
      if (kbase + 16 + col > qrow) v1 = -__builtin_inff();
      s0[r] = v0; s1[r] = v1;
      mb[r] = fmaxf(v0, v1);
    }
#pragma unroll
    for (int off = 1; off < 16; off <<= 1)
#pragma unroll
      for (int r = 0; r < 8; ++r) mb[r] = fmaxf(mb[r], __shfl_xor(mb[r], off, 32));

    // ---- online softmax update ----
    float sc[8], rs[8];
#pragma unroll
    for (int r = 0; r < 8; ++r) {
      float mnew = fmaxf(m_i[r], mb[r]);
      sc[r] = __expf(m_i[r] - mnew);
      m_i[r] = mnew;
      float p0 = __expf(s0[r] - mnew);
      float p1 = __expf(s1[r] - mnew);
      s0[r] = p0; s1[r] = p1;
      rs[r] = p0 + p1;
    }
#pragma unroll
    for (int off = 1; off < 16; off <<= 1)
#pragma unroll
      for (int r = 0; r < 8; ++r) rs[r] += __shfl_xor(rs[r], off, 32);
#pragma unroll
    for (int r = 0; r < 8; ++r) {
      l_i[r] = l_i[r] * sc[r] + rs[r];
      o0[r] *= sc[r]; o1[r] *= sc[r]; o2[r] *= sc[r]; o3[r] *= sc[r];
    }

    // ---- stage P (16x32) to LDS as bf16, re-read in A-fragment order ----
#pragma unroll
    for (int r = 0; r < 8; ++r) {
      int prow = r + 8 * half;
      myP[prow * 32 + col]      = (bf16_t)s0[r];
      myP[prow * 32 + col + 16] = (bf16_t)s1[r];
    }
    asm volatile("s_wait_dscnt 0" ::: "memory");   // order LDS write -> LDS read (same wave)
    v16bf pa = load_frag_a(myP, 0, 32, 0, lane);

    // ---- O += P * V  (V^T stored as [dh, s], contiguous in s) ----
    o0 = wmma_bf16(pa, load_frag_b(Vh, 0,  S_, kbase, lane), o0);
    o1 = wmma_bf16(pa, load_frag_b(Vh, 16, S_, kbase, lane), o1);
    o2 = wmma_bf16(pa, load_frag_b(Vh, 32, S_, kbase, lane), o2);
    o3 = wmma_bf16(pa, load_frag_b(Vh, 48, S_, kbase, lane), o3);
  }

  // ---- normalize and emit to attention buffer [b, s, h*64+dh] (bf16) ----
  v8f os[4] = {o0, o1, o2, o3};
#pragma unroll
  for (int r = 0; r < 8; ++r) {
    float inv = 1.0f / l_i[r];
    int srow = q0 + r + 8 * half;
    size_t base = ((size_t)(b * S_) + srow) * D_ + h * DH;
#pragma unroll
    for (int t = 0; t < 4; ++t)
      Ab[base + t * 16 + col] = (bf16_t)(os[t][r] * inv);
  }
}

// ---------------------------------------------------------------------------
// Launch
// ---------------------------------------------------------------------------
extern "C" void kernel_launch(void* const* d_in, const int* in_sizes, int n_in,
                              void* d_out, int out_size, void* d_ws, size_t ws_size,
                              hipStream_t stream) {
  const float* X   = (const float*)d_in[0];
  const int*   tok = (const int*)d_in[1];
  const float* qw  = (const float*)d_in[2];
  const float* kw  = (const float*)d_in[3];
  const float* vw  = (const float*)d_in[4];
  const float* ow  = (const float*)d_in[5];
  float* out = (float*)d_out;

  const int MROWS = B_ * S_;           // 2048
  char* ws = (char*)d_ws;
  size_t off = 0;
  auto alloc = [&](size_t bytes) { char* p = ws + off; off = (off + bytes + 255) & ~(size_t)255; return p; };

  bf16_t* Xb = (bf16_t*)alloc((size_t)MROWS * D_ * 2);
  bf16_t* Wq = (bf16_t*)alloc((size_t)D_ * D_ * 2);
  bf16_t* Wk = (bf16_t*)alloc((size_t)DKV * D_ * 2);
  bf16_t* Wv = (bf16_t*)alloc((size_t)DKV * D_ * 2);
  bf16_t* Wo = (bf16_t*)alloc((size_t)D_ * D_ * 2);
  float*  Qf = (float*)alloc((size_t)MROWS * D_ * 4);
  float*  Kf = (float*)alloc((size_t)MROWS * DKV * 4);
  float*  Vf = (float*)alloc((size_t)MROWS * DKV * 4);
  bf16_t* Qb = (bf16_t*)alloc((size_t)MROWS * D_ * 2);
  bf16_t* Kb = (bf16_t*)alloc((size_t)MROWS * DKV * 2);
  bf16_t* Vt = (bf16_t*)alloc((size_t)MROWS * DKV * 2);
  bf16_t* Ab = (bf16_t*)alloc((size_t)MROWS * D_ * 2);

  // 1) f32 -> bf16 conversions
  {
    int n;
    n = MROWS * D_;  cvt_bf16_kernel<<<(n + 255) / 256, 256, 0, stream>>>(X,  Xb, n);
    n = D_ * D_;     cvt_bf16_kernel<<<(n + 255) / 256, 256, 0, stream>>>(qw, Wq, n);
    n = DKV * D_;    cvt_bf16_kernel<<<(n + 255) / 256, 256, 0, stream>>>(kw, Wk, n);
    n = DKV * D_;    cvt_bf16_kernel<<<(n + 255) / 256, 256, 0, stream>>>(vw, Wv, n);
    n = D_ * D_;     cvt_bf16_kernel<<<(n + 255) / 256, 256, 0, stream>>>(ow, Wo, n);
  }

  // 2) projections: waves = (M/32)*(N/64), 8 waves per block
  {
    int waves = (MROWS / 32) * (D_ / 64);     // 2048
    gemm_bf16_kernel<<<waves / 8, 256, 0, stream>>>(Xb, Wq, Qf, MROWS, D_, D_);
    waves = (MROWS / 32) * (DKV / 64);        // 512
    gemm_bf16_kernel<<<waves / 8, 256, 0, stream>>>(Xb, Wk, Kf, MROWS, DKV, D_);
    gemm_bf16_kernel<<<waves / 8, 256, 0, stream>>>(Xb, Wv, Vf, MROWS, DKV, D_);
  }

  // 3) RoPE + repack
  {
    int n = B_ * S_ * (D_ / 2);
    rope_pack_q_kernel<<<(n + 255) / 256, 256, 0, stream>>>(Qf, tok, Qb);
    n = B_ * S_ * (DKV / 2);
    rope_pack_k_kernel<<<(n + 255) / 256, 256, 0, stream>>>(Kf, tok, Kb);
    n = B_ * S_ * DKV;
    pack_v_kernel<<<(n + 255) / 256, 256, 0, stream>>>(Vf, Vt);
  }

  // 4) flash attention: B*H*(S/16) waves = 4096 -> 512 blocks of 8 waves
  attn_kernel<<<(B_ * H_ * (S_ / 16)) / 8, 256, 0, stream>>>(Qb, Kb, Vt, Ab);

  // 5) output projection -> f32 d_out
  {
    int waves = (MROWS / 32) * (D_ / 64);
    gemm_bf16_kernel<<<waves / 8, 256, 0, stream>>>(Ab, Wo, out, MROWS, D_, D_);
  }
}
